// Color_ConditionDF_49813030699471
// MI455X (gfx1250) — matrix operand order repaired and checked by reading, not compiled, not run on hardware
//
#include <hip/hip_runtime.h>
#include <hip/hip_bf16.h>

// ---------------------------------------------------------------------------
// Types for CDNA5 WMMA (gfx1250, wave32)
// ---------------------------------------------------------------------------
typedef __bf16 v16bf __attribute__((ext_vector_type(16)));
typedef __bf16 v8bf  __attribute__((ext_vector_type(8)));
typedef float  v8f   __attribute__((ext_vector_type(8)));

#define DDF_K 3
#define TAPS  9
static __device__ __forceinline__ float fn_std() { return 0.47140452079103173f; } // sqrt(2)/3

// ---------------------------------------------------------------------------
// Utility kernels
// ---------------------------------------------------------------------------
__global__ void fill_kernel(float* __restrict__ p, float v, int n) {
    int i = blockIdx.x * blockDim.x + threadIdx.x;
    if (i < n) p[i] = v;
}

__global__ void w2bf_kernel(const float* __restrict__ w, unsigned short* __restrict__ o, int n) {
    int i = blockIdx.x * blockDim.x + threadIdx.x;
    if (i < n) {
        __bf16 h = (__bf16)w[i];                       // native f32->bf16 (RNE)
        o[i] = __builtin_bit_cast(unsigned short, h);
    }
}

// ---------------------------------------------------------------------------
// K1: 1x1 conv as WMMA GEMM.  D[cout,16pix] = W[cout,cin] * X[cin,16pix]
//   - weights staged in LDS, K-padded to KPAD=KT*32 with zeros ->
//     A fragments are two unconditional ds_load_b128 per (mt,kt)
//   - per-(b,cin) affine staged in LDS as float2, zero-padded to KPAD ->
//     B fragment path is branch-free (address clamp + fma with 0 affine)
//   - epilogue: bias add (LDS-staged), store x1, per-lane GAP accumulators;
//     one cross-lane reduce + ds_add_f32 per WG, one global atomic/channel.
// 128 threads = 4 waves; each wave does 4 tiles of 16 pixels (256 pix / WG).
// HW is guaranteed to be a multiple of 256 (min 32x32=1024) -> no tail guard.
// ---------------------------------------------------------------------------
template<int CIN, int COUT>
__global__ __launch_bounds__(128)
void conv1x1_wmma_kernel(const float* __restrict__ xin,
                         const unsigned short* __restrict__ wbf,
                         const float* __restrict__ bias,
                         const float* __restrict__ affA,
                         const float* __restrict__ affB,
                         float* __restrict__ xout,
                         float* __restrict__ gap,
                         int HW)
{
    constexpr int KT   = (CIN + 31) / 32;   // K tiles of 32 (bf16 WMMA K)
    constexpr int KPAD = KT * 32;           // zero-padded K extent (power of 2)
    constexpr int MT   = COUT / 16;         // M tiles of 16 (cout)
    constexpr int KSH  = (KPAD == 32) ? 5 : (KPAD == 64 ? 6 : 7);
    static_assert(COUT * KPAD <= 16384, "weights must fit LDS stage");

    __shared__ unsigned short wlds[COUT * KPAD];   // zero-padded weights (bf16 bits)
    __shared__ float2 aff_l[KPAD];                 // zero-padded (scale, shift)
    __shared__ float bias_l[COUT];
    __shared__ float gap_l[COUT];

    const int tid = threadIdx.x;
    const int b   = blockIdx.z;

    for (int i = tid; i < COUT * KPAD; i += 128) {
        const int row = i >> KSH;
        const int k   = i & (KPAD - 1);
        wlds[i] = (k < CIN) ? wbf[row * CIN + k] : (unsigned short)0;
    }
    if (tid < COUT) { bias_l[tid] = bias[tid]; gap_l[tid] = 0.f; }
    if (tid < KPAD) {
        float2 ab = make_float2(0.f, 0.f);
        if (tid < CIN) ab = make_float2(affA[b * CIN + tid], affB[b * CIN + tid]);
        aff_l[tid] = ab;
    }
    __syncthreads();

    const int wave = tid >> 5;
    const int lane = tid & 31;
    const int nIdx = lane & 15;   // column (pixel) within tile
    const int hi   = lane >> 4;   // half-wave selector
    const size_t inB  = (size_t)b * CIN  * (size_t)HW;
    const size_t outB = (size_t)b * COUT * (size_t)HW;

    // per-lane GAP accumulators, one set of 8 rows per M-tile
    float gacc[MT][8];
    #pragma unroll
    for (int mt = 0; mt < MT; ++mt)
        #pragma unroll
        for (int r = 0; r < 8; ++r) gacc[mt][r] = 0.f;

    #pragma unroll 1
    for (int t = 0; t < 4; ++t) {
        const int p = blockIdx.x * 256 + (wave * 4 + t) * 16 + nIdx;

        // ---- B fragments: X[cin, 16 pixels] in bf16, contiguous k order, same as A:
        // lane L holds column n=L&15, K = (L>>4)*16 + i.  Branch-free:
        // out-of-range K reads channel 0 but multiplies by zero affine.
        v16bf bfrag[KT];
        #pragma unroll
        for (int kt = 0; kt < KT; ++kt) {
            #pragma unroll
            for (int i = 0; i < 16; ++i) {
                const int kk = kt * 32 + hi * 16 + i;
                const int kc = (kk < CIN) ? kk : 0;          // v_cndmask, no branch
                const float2 ab = aff_l[kk];                 // single ds_load_b64
                const float v = fmaf(xin[inB + (size_t)kc * HW + p], ab.x, ab.y);
                bfrag[kt][i] = (__bf16)v;                    // native cvt / fma_mix
            }
        }

        #pragma unroll
        for (int mt = 0; mt < MT; ++mt) {
            v8f acc = {};
            const int row = mt * 16 + nIdx;   // A: lane L holds row m=L&15
            #pragma unroll
            for (int kt = 0; kt < KT; ++kt) {
                // A layout: elements 0..15 at K=base..base+15 (base = kt*32 + hi*16)
                const v8bf* wv = (const v8bf*)&wlds[row * KPAD + kt * 32 + hi * 16];
                const v8bf alo = wv[0];      // ds_load_b128
                const v8bf ahi = wv[1];      // ds_load_b128 (+8 shorts)
                const v16bf afrag = __builtin_shufflevector(
                    alo, ahi, 0, 1, 2, 3, 4, 5, 6, 7, 8, 9, 10, 11, 12, 13, 14, 15);
                acc = __builtin_amdgcn_wmma_f32_16x16x32_bf16(
                          false, afrag, false, bfrag[kt], (short)0, acc, false, false);
            }

            // ---- epilogue: bias + store + per-lane GAP accumulation
            #pragma unroll
            for (int r = 0; r < 8; ++r) {
                const int co = mt * 16 + hi * 8 + r;   // C/D layout row
                const float v = acc[r] + bias_l[co];
                xout[outB + (size_t)co * HW + p] = v;
                gacc[mt][r] += v;
            }
        }
    }

    // ---- once per workgroup: cross-lane reduce (within the 16 pixel-lanes
    // of each half-wave), then LDS float atomic, then one global atomic/chan.
    #pragma unroll
    for (int mt = 0; mt < MT; ++mt) {
        #pragma unroll
        for (int off = 1; off < 16; off <<= 1) {
            #pragma unroll
            for (int r = 0; r < 8; ++r) gacc[mt][r] += __shfl_xor(gacc[mt][r], off, 32);
        }
        if (nIdx == 0) {
            #pragma unroll
            for (int r = 0; r < 8; ++r)
                atomicAdd(&gap_l[mt * 16 + hi * 8 + r], gacc[mt][r]);  // ds_add_f32
        }
    }
    __syncthreads();
    for (int i = tid; i < COUT; i += 128) atomicAdd(&gap[b * COUT + i], gap_l[i]);
}

// ---------------------------------------------------------------------------
// K2: spatial filter branch. sp[b,9,p] = FilterNorm( sp_w @ x1[:,p] + sp_b )
// ---------------------------------------------------------------------------
__global__ __launch_bounds__(256)
void sp_kernel(const float* __restrict__ x1,
               const float* __restrict__ sp_w,   // [9, cout]
               const float* __restrict__ sp_b,   // [9]
               float* __restrict__ sp,           // [b,9,HW]
               int cout, int HW)
{
    __shared__ float w[TAPS * 128];
    const int tid = threadIdx.x;
    const int b   = blockIdx.z;
    for (int i = tid; i < TAPS * cout; i += 256) w[i] = sp_w[i];
    __syncthreads();

    const int p = blockIdx.x * 256 + tid;

    float acc[TAPS];
    #pragma unroll
    for (int j = 0; j < TAPS; ++j) acc[j] = sp_b[j];

    const float* xb = x1 + (size_t)b * cout * (size_t)HW + p;
    for (int c = 0; c < cout; ++c) {
        if (c + 16 < cout) __builtin_prefetch(xb + (size_t)(c + 16) * HW);  // global_prefetch_b8
        const float v = xb[(size_t)c * HW];
        #pragma unroll
        for (int j = 0; j < TAPS; ++j) acc[j] = fmaf(w[j * cout + c], v, acc[j]);
    }

    float m = 0.f;
    #pragma unroll
    for (int j = 0; j < TAPS; ++j) m += acc[j];
    m *= (1.f / 9.f);
    float var = 0.f;
    #pragma unroll
    for (int j = 0; j < TAPS; ++j) { float d = acc[j] - m; var += d * d; }
    var *= (1.f / 8.f);                       // unbiased (ddof=1)
    const float sc = fn_std() / (sqrtf(var) + 1e-10f);

    float* spo = sp + (size_t)b * TAPS * (size_t)HW + p;
    #pragma unroll
    for (int j = 0; j < TAPS; ++j) spo[(size_t)j * HW] = (acc[j] - m) * sc;
}

// ---------------------------------------------------------------------------
// K3: channel branch (tiny). GAP -> SE bottleneck -> FilterNorm per channel.
// One workgroup per batch.
// ---------------------------------------------------------------------------
__global__ __launch_bounds__(128)
void ch_kernel(const float* __restrict__ gap,   // [b,cout] (sums)
               const float* __restrict__ w1,    // [mid, cout]
               const float* __restrict__ b1,    // [mid]
               const float* __restrict__ w2,    // [cout*9, mid]
               const float* __restrict__ b2,    // [cout*9]
               float* __restrict__ ch,          // [b,cout,9]
               int cout, int mid, int HW)
{
    __shared__ float g[128];
    __shared__ float h1[32];
    const int b = blockIdx.x;
    const int t = threadIdx.x;

    if (t < cout) g[t] = gap[b * cout + t] / (float)HW;
    __syncthreads();
    if (t < mid) {
        float a = b1[t];
        for (int c = 0; c < cout; ++c) a = fmaf(w1[t * cout + c], g[c], a);
        h1[t] = fmaxf(a, 0.f);
    }
    __syncthreads();
    if (t < cout) {
        float r[TAPS];
        #pragma unroll
        for (int j = 0; j < TAPS; ++j) {
            float a = b2[t * TAPS + j];
            for (int m2 = 0; m2 < mid; ++m2) a = fmaf(w2[(t * TAPS + j) * mid + m2], h1[m2], a);
            r[j] = a;
        }
        float m = 0.f;
        #pragma unroll
        for (int j = 0; j < TAPS; ++j) m += r[j];
        m *= (1.f / 9.f);
        float var = 0.f;
        #pragma unroll
        for (int j = 0; j < TAPS; ++j) { float d = r[j] - m; var += d * d; }
        var *= (1.f / 8.f);
        const float sc = fn_std() / (sqrtf(var) + 1e-10f);
        float* cho = ch + ((size_t)b * cout + t) * TAPS;
        #pragma unroll
        for (int j = 0; j < TAPS; ++j) cho[j] = (r[j] - m) * sc;
    }
}

// ---------------------------------------------------------------------------
// K4: DDF(mul) + AvgPool(3,s2,p1,count_include_pad) + LeakyReLU(0.2)
//     + fused instance-norm sum/sumsq reduction.
// Thread = one pooled output pixel; grid = (pix tiles, c, b).
// ---------------------------------------------------------------------------
__global__ __launch_bounds__(256)
void ddf_pool_kernel(const float* __restrict__ x1,   // [b,c,H,W]
                     const float* __restrict__ sp,   // [b,9,H,W]
                     const float* __restrict__ ch,   // [b,c,9]
                     float* __restrict__ y,          // [b,c,H/2,W/2]
                     float* __restrict__ stats,      // [b,c,2]
                     int cout, int H, int W, int do_stats)
{
    __shared__ float rs[256];
    __shared__ float rq[256];

    const int b  = blockIdx.z;
    const int c  = blockIdx.y;
    const int H2 = H >> 1, W2 = W >> 1;
    const int tid = threadIdx.x;
    const int idx = blockIdx.x * 256 + tid;

    float f[TAPS];
    const float* chp = ch + ((size_t)b * cout + c) * TAPS;
    #pragma unroll
    for (int j = 0; j < TAPS; ++j) f[j] = chp[j];

    float yv = 0.f;
    const bool valid = idx < H2 * W2;
    if (valid) {
        const int i2 = idx / W2, j2 = idx % W2;
        const float* xb  = x1 + ((size_t)b * cout + c) * (size_t)H * W;
        const float* spb = sp + (size_t)b * TAPS * (size_t)H * W;
        float acc = 0.f;
        #pragma unroll
        for (int a = 0; a < 3; ++a) {
            const int r = 2 * i2 - 1 + a;
            if (r < 0 || r >= H) continue;
            #pragma unroll
            for (int q = 0; q < 3; ++q) {
                const int s = 2 * j2 - 1 + q;
                if (s < 0 || s >= W) continue;
                float z = 0.f;
                #pragma unroll
                for (int u = 0; u < 3; ++u) {
                    const int rr = r + u - 1;
                    if (rr < 0 || rr >= H) continue;
                    #pragma unroll
                    for (int v = 0; v < 3; ++v) {
                        const int ss = s + v - 1;
                        if (ss < 0 || ss >= W) continue;
                        z = fmaf(xb[(size_t)rr * W + ss] * f[u * 3 + v],
                                 spb[(size_t)(u * 3 + v) * H * W + (size_t)r * W + s], z);
                    }
                }
                acc += z;
            }
        }
        yv = acc * (1.f / 9.f);
        yv = yv > 0.f ? yv : 0.2f * yv;
        y[((size_t)b * cout + c) * (size_t)(H2 * W2) + idx] = yv;
    }

    if (do_stats) {
        rs[tid] = valid ? yv : 0.f;
        rq[tid] = valid ? yv * yv : 0.f;
        __syncthreads();
        for (int s = 128; s > 0; s >>= 1) {
            if (tid < s) { rs[tid] += rs[tid + s]; rq[tid] += rq[tid + s]; }
            __syncthreads();
        }
        if (tid == 0) {
            atomicAdd(&stats[((size_t)b * cout + c) * 2 + 0], rs[0]);
            atomicAdd(&stats[((size_t)b * cout + c) * 2 + 1], rq[0]);
        }
    }
}

// ---------------------------------------------------------------------------
// K5: instance-norm finalize -> per-(b,c) affine consumed by next conv.
// ---------------------------------------------------------------------------
__global__ void norm_fin_kernel(const float* __restrict__ stats,
                                const float* __restrict__ gamma,
                                const float* __restrict__ beta,
                                float* __restrict__ affA, float* __restrict__ affB,
                                int cout, int cnt)
{
    const int i = blockIdx.x * blockDim.x + threadIdx.x;
    if (i >= 8 * cout) return;
    const int c = i % cout;
    const float inv = 1.f / (float)cnt;
    const float m = stats[(size_t)i * 2 + 0] * inv;
    const float v = stats[(size_t)i * 2 + 1] * inv - m * m;   // biased var
    const float a = gamma[c] * rsqrtf(v + 1e-5f);
    affA[i] = a;
    affB[i] = beta[c] - m * a;
}

// ---------------------------------------------------------------------------
// K6: head: 1x1 conv 128->3 on [8,128,16,16] + global mean -> [8,3,1,1]
// ---------------------------------------------------------------------------
__global__ __launch_bounds__(256)
void head_kernel(const float* __restrict__ y,     // [b,128,256]
                 const float* __restrict__ ow,    // [3,128]
                 const float* __restrict__ ob,    // [3]
                 float* __restrict__ out)         // [b,3]
{
    __shared__ float red[256];
    const int b = blockIdx.x, tid = threadIdx.x;
    const float* yb = y + (size_t)b * 128 * 256;
    float o[3] = {0.f, 0.f, 0.f};
    for (int c = 0; c < 128; ++c) {
        const float v = yb[(size_t)c * 256 + tid];
        o[0] = fmaf(ow[c],        v, o[0]);
        o[1] = fmaf(ow[128 + c],  v, o[1]);
        o[2] = fmaf(ow[256 + c],  v, o[2]);
    }
    #pragma unroll
    for (int k = 0; k < 3; ++k) {
        red[tid] = o[k];
        __syncthreads();
        for (int s = 128; s > 0; s >>= 1) {
            if (tid < s) red[tid] += red[tid + s];
            __syncthreads();
        }
        if (tid == 0) out[b * 3 + k] = red[0] * (1.f / 256.f) + ob[k];
        __syncthreads();
    }
}

// ---------------------------------------------------------------------------
// Host orchestration
// ---------------------------------------------------------------------------
extern "C" void kernel_launch(void* const* d_in, const int* in_sizes, int n_in,
                              void* d_out, int out_size, void* d_ws, size_t ws_size,
                              hipStream_t stream) {
    (void)n_in; (void)out_size; (void)ws_size;

    static const int cinA[5]  = {3, 16, 32, 64, 128};
    static const int coutA[5] = {16, 32, 64, 128, 128};
    static const int midA[5]  = {3, 6, 12, 25, 25};
    static const int normA[5] = {1, 1, 1, 1, 0};
    static const int HinA[5]  = {512, 256, 128, 64, 32};

    const float* img = (const float*)d_in[0];

    struct BP { const float *cw, *cb, *sw, *sb, *w1, *b1, *w2, *b2, *ga, *be; } bp[5];
    const float *out_w, *out_b;

    // pytree order detection: insertion (conv_w first, 48 elems) vs
    // alphabetical (beta first, 16 elems) for block 0.
    const bool alpha = (in_sizes[1] == 16);
    int idx = 1;
    for (int i = 0; i < 5; ++i) {
        if (!alpha) {
            bp[i].cw = (const float*)d_in[idx++]; bp[i].cb = (const float*)d_in[idx++];
            bp[i].sw = (const float*)d_in[idx++]; bp[i].sb = (const float*)d_in[idx++];
            bp[i].w1 = (const float*)d_in[idx++]; bp[i].b1 = (const float*)d_in[idx++];
            bp[i].w2 = (const float*)d_in[idx++]; bp[i].b2 = (const float*)d_in[idx++];
            if (normA[i]) { bp[i].ga = (const float*)d_in[idx++]; bp[i].be = (const float*)d_in[idx++]; }
            else          { bp[i].ga = nullptr; bp[i].be = nullptr; }
        } else {
            if (normA[i]) bp[i].be = (const float*)d_in[idx++];
            bp[i].b1 = (const float*)d_in[idx++]; bp[i].b2 = (const float*)d_in[idx++];
            bp[i].w1 = (const float*)d_in[idx++]; bp[i].w2 = (const float*)d_in[idx++];
            bp[i].cb = (const float*)d_in[idx++]; bp[i].cw = (const float*)d_in[idx++];
            if (normA[i]) bp[i].ga = (const float*)d_in[idx++]; else { bp[i].ga = nullptr; bp[i].be = nullptr; }
            bp[i].sb = (const float*)d_in[idx++]; bp[i].sw = (const float*)d_in[idx++];
        }
    }
    if (!alpha) { out_w = (const float*)d_in[idx++]; out_b = (const float*)d_in[idx++]; }
    else        { out_b = (const float*)d_in[idx++]; out_w = (const float*)d_in[idx++]; }

    // workspace carve-up
    char* ws = (char*)d_ws;
    size_t off = 0;
    auto alloc = [&](size_t bytes) -> void* {
        void* p = ws + off;
        off = (off + bytes + 255) & ~(size_t)255;
        return p;
    };
    unsigned short* wbf[5];
    for (int i = 0; i < 5; ++i) wbf[i] = (unsigned short*)alloc((size_t)coutA[i] * cinA[i] * 2);
    float* gap   = (float*)alloc(8 * 128 * 4);
    float* chb   = (float*)alloc(8 * 128 * TAPS * 4);
    float* stats = (float*)alloc(8 * 128 * 2 * 4);
    float* affA  = (float*)alloc(8 * 128 * 4);
    float* affB  = (float*)alloc(8 * 128 * 4);
    float* x1    = (float*)alloc((size_t)8 * 16 * 512 * 512 * 4);   // 134 MB (max x1)
    float* spb   = (float*)alloc((size_t)8 * TAPS * 512 * 512 * 4); // 75.5 MB (max sp)
    float* y0    = (float*)alloc((size_t)8 * 16 * 256 * 256 * 4);   // 33.5 MB (max y)
    float* y1    = (float*)alloc((size_t)8 * 16 * 256 * 256 * 4);

    // weights -> bf16 (once per launch; deterministic)
    for (int i = 0; i < 5; ++i) {
        const int n = coutA[i] * cinA[i];
        w2bf_kernel<<<(n + 255) / 256, 256, 0, stream>>>(bp[i].cw, wbf[i], n);
    }
    // identity affine for block 0 (conv always applies affine -> branch-free)
    fill_kernel<<<(8 * 128 + 255) / 256, 256, 0, stream>>>(affA, 1.f, 8 * 128);
    fill_kernel<<<(8 * 128 + 255) / 256, 256, 0, stream>>>(affB, 0.f, 8 * 128);

    const float* cur = img;
    float* ybufs[2] = {y0, y1};
    for (int i = 0; i < 5; ++i) {
        const int cout = coutA[i], H = HinA[i], W = H, HW = H * W;
        const int H2 = H >> 1, W2 = W >> 1;

        fill_kernel<<<(8 * cout + 255) / 256, 256, 0, stream>>>(gap, 0.f, 8 * cout);
        if (normA[i])
            fill_kernel<<<(8 * cout * 2 + 255) / 256, 256, 0, stream>>>(stats, 0.f, 8 * cout * 2);

        dim3 gConv(HW / 256, 1, 8);
        switch (i) {
        case 0: conv1x1_wmma_kernel<3, 16><<<gConv, 128, 0, stream>>>(cur, wbf[0], bp[0].cb, affA, affB, x1, gap, HW); break;
        case 1: conv1x1_wmma_kernel<16, 32><<<gConv, 128, 0, stream>>>(cur, wbf[1], bp[1].cb, affA, affB, x1, gap, HW); break;
        case 2: conv1x1_wmma_kernel<32, 64><<<gConv, 128, 0, stream>>>(cur, wbf[2], bp[2].cb, affA, affB, x1, gap, HW); break;
        case 3: conv1x1_wmma_kernel<64, 128><<<gConv, 128, 0, stream>>>(cur, wbf[3], bp[3].cb, affA, affB, x1, gap, HW); break;
        case 4: conv1x1_wmma_kernel<128, 128><<<gConv, 128, 0, stream>>>(cur, wbf[4], bp[4].cb, affA, affB, x1, gap, HW); break;
        }

        sp_kernel<<<dim3(HW / 256, 1, 8), 256, 0, stream>>>(x1, bp[i].sw, bp[i].sb, spb, cout, HW);
        ch_kernel<<<8, 128, 0, stream>>>(gap, bp[i].w1, bp[i].b1, bp[i].w2, bp[i].b2, chb, cout, midA[i], HW);

        float* y = ybufs[i & 1];
        ddf_pool_kernel<<<dim3((H2 * W2 + 255) / 256, cout, 8), 256, 0, stream>>>(
            x1, spb, chb, y, stats, cout, H, W, normA[i]);

        if (normA[i]) {
            norm_fin_kernel<<<(8 * cout + 255) / 256, 256, 0, stream>>>(
                stats, bp[i].ga, bp[i].be, affA, affB, cout, H2 * W2);
        }
        cur = y;
    }

    head_kernel<<<8, 256, 0, stream>>>(cur, out_w, out_b, (float*)d_out);
}
